// SparseDecoderLayer_67491116089604
// MI455X (gfx1250) — matrix operand order repaired and checked by reading, not compiled, not run on hardware
//
#include <hip/hip_runtime.h>
#include <cstdint>
#include <cstddef>

#define BATCH 2048
#define IN_F  8192
#define OUT_F 8192
#define NNZ   4194304

typedef __attribute__((ext_vector_type(16))) __bf16 v16bf;
typedef __attribute__((ext_vector_type(8)))  __bf16 v8bf;
typedef __attribute__((ext_vector_type(8)))  float  v8f;

__device__ __forceinline__ unsigned short f32_to_bf16(float f) {
    unsigned int u = __float_as_uint(f);
    unsigned int r = 0x7FFFu + ((u >> 16) & 1u);   // round-to-nearest-even
    return (unsigned short)((u + r) >> 16);
}

// ---------------------------------------------------------------- zero W (bf16)
__global__ void __launch_bounds__(256) zero_kernel(uint4* __restrict__ p) {
    size_t i = (size_t)blockIdx.x * 256 + threadIdx.x;
    p[i] = make_uint4(0u, 0u, 0u, 0u);
}

// ---------------------------------------------------------------- x f32 -> bf16
__global__ void __launch_bounds__(256) cvt_x_kernel(const float* __restrict__ x,
                                                    unsigned short* __restrict__ xb) {
    size_t i = ((size_t)blockIdx.x * 256 + threadIdx.x) * 8;
    float4 a = *(const float4*)(x + i);
    float4 b = *(const float4*)(x + i + 4);
    uint4 o;
    o.x = (unsigned)f32_to_bf16(a.x) | ((unsigned)f32_to_bf16(a.y) << 16);
    o.y = (unsigned)f32_to_bf16(a.z) | ((unsigned)f32_to_bf16(a.w) << 16);
    o.z = (unsigned)f32_to_bf16(b.x) | ((unsigned)f32_to_bf16(b.y) << 16);
    o.w = (unsigned)f32_to_bf16(b.z) | ((unsigned)f32_to_bf16(b.w) << 16);
    *(uint4*)(xb + i) = o;
}

// ------------------------------------------------- COO scatter with pk bf16 atomics
__global__ void __launch_bounds__(256) scatter_kernel(const long long* __restrict__ idx,
                                                      const float* __restrict__ vals,
                                                      unsigned int* __restrict__ wb) {
    size_t i = (size_t)blockIdx.x * 256 + threadIdx.x;
    long long r = idx[i];
    long long c = idx[i + (size_t)NNZ];
    size_t e = (size_t)r * IN_F + (size_t)c;          // element index into W[o][k]
    unsigned short bv = f32_to_bf16(vals[i]);
    unsigned int pk = (e & 1) ? ((unsigned int)bv << 16) : (unsigned int)bv; // other half = +0.0
    unsigned int* addr = wb + (e >> 1);
    asm volatile("global_atomic_pk_add_bf16 %0, %1, off scope:SCOPE_DEV"
                 :: "v"(addr), "v"(pk) : "memory");
}

// -------------------------------------------------------- async global -> LDS copy
// GVS mode: lds[VDST + b] = mem[SADDR + VADDR + b], 16 bytes/lane, tracked by ASYNCcnt
__device__ __forceinline__ void async_copy16(unsigned lds_byte_addr, unsigned voff_bytes,
                                             const void* sbase) {
    asm volatile("global_load_async_to_lds_b128 %0, %1, %2"
                 :: "v"(lds_byte_addr), "v"(voff_bytes), "s"(sbase) : "memory");
}

// ---------------------------------------------------------------- tiled WMMA GEMM
// C[b,o] = sum_k Xb[b,k] * Wb[o,k] ; Xb: [2048,8192] bf16, Wb: [8192,8192] bf16
// Block tile 128(M) x 256(N), 8 waves (2x4), 64x64 per wave = 4x4 WMMA tiles.
#define KSTEP 64
#define LDSR  72    // 64 halfs + 8 pad = 144B row stride (16B-aligned, spreads banks)

__global__ void __launch_bounds__(256)
gemm_kernel(const unsigned short* __restrict__ Xb,
            const unsigned short* __restrict__ Wb,
            float* __restrict__ out) {
    __shared__ __align__(16) unsigned short ldsA[2][128 * LDSR];
    __shared__ __align__(16) unsigned short ldsB[2][256 * LDSR];

    const int tid  = threadIdx.x;
    const int lane = tid & 31;
    const int wave = tid >> 5;       // 8 waves, 2 (M) x 4 (N)
    const int sel  = lane >> 4;      // lane-half select (ISA fragment layout)
    const int l16  = lane & 15;
    const int wm   = wave & 1;       // 64 M-rows each
    const int wn   = wave >> 1;      // 64 N-cols each

    const int m0 = blockIdx.y * 128;
    const int n0 = blockIdx.x * 256;

    // staging: A tile 128x64 halfs = 1024 x 16B chunks (4/thread)
    //          B tile 256x64 halfs = 2048 x 16B chunks (8/thread)
    unsigned voffA[4], ldsDstA[4], voffB[8], ldsDstB[8];
    const unsigned baseA = (unsigned)(unsigned long long)(const void*)&ldsA[0][0];
    const unsigned baseB = (unsigned)(unsigned long long)(const void*)&ldsB[0][0];
#pragma unroll
    for (int t = 0; t < 4; ++t) {
        int c = tid + 256 * t;
        int row = c >> 3, cc = (c & 7) * 8;           // 8 chunks per 64-half row
        voffA[t]   = (unsigned)(((m0 + row) * IN_F + cc) * 2);
        ldsDstA[t] = baseA + (unsigned)((row * LDSR + cc) * 2);
    }
#pragma unroll
    for (int t = 0; t < 8; ++t) {
        int c = tid + 256 * t;
        int row = c >> 3, cc = (c & 7) * 8;
        voffB[t]   = (unsigned)(((n0 + row) * IN_F + cc) * 2);
        ldsDstB[t] = baseB + (unsigned)((row * LDSR + cc) * 2);
    }
    const unsigned BUFA = 128 * LDSR * 2;             // bytes per A buffer
    const unsigned BUFB = 256 * LDSR * 2;             // bytes per B buffer

    v8f acc[4][4] = {};

    // prologue: async-stage K-slab 0 into buffer 0
#pragma unroll
    for (int t = 0; t < 4; ++t) async_copy16(ldsDstA[t], voffA[t], Xb);
#pragma unroll
    for (int t = 0; t < 8; ++t) async_copy16(ldsDstB[t], voffB[t], Wb);
    asm volatile("s_wait_asynccnt 0" ::: "memory");
    __syncthreads();

    const int NK = IN_F / KSTEP;    // 128
    for (int kb = 0; kb < NK; ++kb) {
        const int cur = kb & 1;
        const int nxt = cur ^ 1;

        if (kb + 1 < NK) {          // async copy of next slab, overlapped with WMMAs
            const unsigned short* sX = Xb + (size_t)(kb + 1) * KSTEP;
            const unsigned short* sW = Wb + (size_t)(kb + 1) * KSTEP;
#pragma unroll
            for (int t = 0; t < 4; ++t) async_copy16(ldsDstA[t] + (unsigned)nxt * BUFA, voffA[t], sX);
#pragma unroll
            for (int t = 0; t < 8; ++t) async_copy16(ldsDstB[t] + (unsigned)nxt * BUFB, voffB[t], sW);
        }

        const unsigned short* As = &ldsA[cur][0];
        const unsigned short* Bs = &ldsB[cur][0];

#pragma unroll
        for (int kk = 0; kk < 2; ++kk) {              // two K=32 WMMA slices per slab
            v16bf afrag[4], bfrag[4];
#pragma unroll
            for (int j = 0; j < 4; ++j) {             // B: lane = column n, 16 contig K halfs
                const unsigned short* p = Bs + (wn * 64 + j * 16 + l16) * LDSR + kk * 32 + sel * 16;
                v8bf lo = *(const v8bf*)(p);
                v8bf hi = *(const v8bf*)(p + 8);
                bfrag[j] = __builtin_shufflevector(lo, hi, 0,1,2,3,4,5,6,7,8,9,10,11,12,13,14,15);
            }
#pragma unroll
            for (int i = 0; i < 4; ++i) {             // A: lane = row m, 16 contig K halfs
                const unsigned short* p = As + (wm * 64 + i * 16 + l16) * LDSR + kk * 32 + sel * 16;
                v8bf lo = *(const v8bf*)(p);
                v8bf hi = *(const v8bf*)(p + 8);
                afrag[i] = __builtin_shufflevector(lo, hi, 0,1,2,3,4,5,6,7,8,9,10,11,12,13,14,15);
            }
#pragma unroll
            for (int i = 0; i < 4; ++i)
#pragma unroll
                for (int j = 0; j < 4; ++j)
                    acc[i][j] = __builtin_amdgcn_wmma_f32_16x16x32_bf16(
                        false, afrag[i], false, bfrag[j], (short)0, acc[i][j], false, false);
        }

        asm volatile("s_wait_asynccnt 0" ::: "memory");   // our slab-copy done
        __syncthreads();                                   // everyone's copy visible
    }

    // epilogue: C/D layout -> lane = column n, VGPR r = row (sel*8 + r); NT stores
#pragma unroll
    for (int i = 0; i < 4; ++i) {
#pragma unroll
        for (int j = 0; j < 4; ++j) {
            int ncol = n0 + wn * 64 + j * 16 + l16;
#pragma unroll
            for (int r = 0; r < 8; ++r) {
                int mrow = m0 + wm * 64 + i * 16 + sel * 8 + r;
                __builtin_nontemporal_store(acc[i][j][r], &out[(size_t)mrow * OUT_F + ncol]);
            }
        }
    }
}

extern "C" void kernel_launch(void* const* d_in, const int* in_sizes, int n_in,
                              void* d_out, int out_size, void* d_ws, size_t ws_size,
                              hipStream_t stream) {
    const float*      x       = (const float*)d_in[0];
    const float*      values  = (const float*)d_in[1];
    const long long*  indices = (const long long*)d_in[2];   // int64, [2, NNZ] flat

    // ws layout: [0, 128MB) W bf16 ; [128MB, 160MB) X bf16
    unsigned short* Wb = (unsigned short*)d_ws;
    unsigned short* Xb = (unsigned short*)((char*)d_ws + (size_t)OUT_F * IN_F * 2);

    // 1) zero W (128MB = 8M uint4)
    zero_kernel<<<32768, 256, 0, stream>>>((uint4*)Wb);
    // 2) convert x -> bf16 (16M floats, 8 per thread)
    cvt_x_kernel<<<8192, 256, 0, stream>>>(x, Xb);
    // 3) COO scatter with packed bf16 atomics
    scatter_kernel<<<NNZ / 256, 256, 0, stream>>>(indices, values, (unsigned int*)Wb);
    // 4) WMMA GEMM: out[2048,8192] = Xb @ Wb^T
    gemm_kernel<<<dim3(OUT_F / 256, BATCH / 128), 256, 0, stream>>>(Xb, Wb, (float*)d_out);
}
